// TweetAugmentedRGCN_5531917877295
// MI455X (gfx1250) — matrix operand order repaired and checked by reading, not compiled, not run on hardware
//
#include <hip/hip_runtime.h>
#include <hip/hip_bf16.h>
#include <stdint.h>

#define N_USERS  100000
#define N_TWEETS 100000
#define N_NODES  200000
#define N_EDGES  600000
#define D        128

typedef __attribute__((ext_vector_type(16))) __bf16 v16bf;
typedef __attribute__((ext_vector_type(8)))  __bf16 v8bf;
typedef __attribute__((ext_vector_type(8)))  float  v8f;

__device__ __forceinline__ float lrelu(float x) { return x > 0.0f ? x : 0.01f * x; }

__device__ __forceinline__ v16bf cat8(v8bf lo, v8bf hi) {
  return __builtin_shufflevector(lo, hi, 0, 1, 2, 3, 4, 5, 6, 7,
                                 8, 9, 10, 11, 12, 13, 14, 15);
}

// ---------------------------------------------------------------------------
// Weight prep: f32 -> bf16 once per launch; RGCN mats transposed to [n][k]
// so every WMMA B-fragment load is a contiguous 16B/32B run per lane.
// ---------------------------------------------------------------------------
__global__ void cvt_rowmajor(const float* __restrict__ src, __bf16* __restrict__ dst, int n) {
  int i = blockIdx.x * blockDim.x + threadIdx.x;
  if (i < n) dst[i] = (__bf16)src[i];
}

__global__ void cvt_transpose128(const float* __restrict__ src /*[k][n]*/,
                                 __bf16* __restrict__ dst /*[n][k]*/) {
  int i = blockIdx.x * blockDim.x + threadIdx.x;   // 16384
  int k = i & 127, n = i >> 7;
  dst[(size_t)n * D + k] = (__bf16)src[(size_t)k * D + n];
}

// ---------------------------------------------------------------------------
// Feature encoders: one 128-thread block per node, input row staged in LDS.
// f32 math, bf16 output.
// ---------------------------------------------------------------------------
__global__ void encode_users(const float* __restrict__ des,
                             const float* __restrict__ num_prop,
                             const float* __restrict__ cat_prop,
                             const float* __restrict__ W_des, const float* __restrict__ b_des,
                             const float* __restrict__ W_num, const float* __restrict__ b_num,
                             const float* __restrict__ W_cat, const float* __restrict__ b_cat,
                             __bf16* __restrict__ X) {
  int u = blockIdx.x;
  int t = threadIdx.x;
  __shared__ float s_des[100];
  __shared__ float s_num[6];
  __shared__ float s_cat[11];
  if (t < 100)      s_des[t]       = des[(size_t)u * 100 + t];
  else if (t < 106) s_num[t - 100] = num_prop[(size_t)u * 6 + (t - 100)];
  else if (t < 117) s_cat[t - 106] = cat_prop[(size_t)u * 11 + (t - 106)];
  __syncthreads();

  float acc;
  if (t < 64) {
    acc = b_des[t];
    const float* w = W_des + (size_t)t * 100;
    #pragma unroll 4
    for (int k = 0; k < 100; ++k) acc += w[k] * s_des[k];
  } else if (t < 96) {
    int j = t - 64;
    acc = b_num[j];
    const float* w = W_num + (size_t)j * 6;
    #pragma unroll
    for (int k = 0; k < 6; ++k) acc += w[k] * s_num[k];
  } else {
    int j = t - 96;
    acc = b_cat[j];
    const float* w = W_cat + (size_t)j * 11;
    #pragma unroll
    for (int k = 0; k < 11; ++k) acc += w[k] * s_cat[k];
  }
  X[(size_t)u * D + t] = (__bf16)lrelu(acc);
}

__global__ void encode_tweets(const float* __restrict__ tweet,
                              const float* __restrict__ W_tw, const float* __restrict__ b_tw,
                              __bf16* __restrict__ Xt) {
  int u = blockIdx.x;
  int t = threadIdx.x;
  __shared__ float srow[100];
  if (t < 100) srow[t] = tweet[(size_t)u * 100 + t];
  __syncthreads();
  float acc = b_tw[t];
  const float* w = W_tw + (size_t)t * 100;
  #pragma unroll 4
  for (int k = 0; k < 100; ++k) acc += w[k] * srow[k];
  Xt[(size_t)u * D + t] = (__bf16)lrelu(acc);
}

// ---------------------------------------------------------------------------
// WMMA fragment loaders (bf16 storage, contiguous per-lane runs):
//   A lane layout: m=lane&15, half=lane>>4; element i<8  -> K = k0+half*8+i,
//                  element 8+i -> K = k0+16+half*8+i.
//   B lane layout: n=lane&15, half=lane>>4; element i<8  -> K = k0+half*8+i,
//                  element 8+i -> K = k0+16+half*8+i.
// ---------------------------------------------------------------------------
__device__ __forceinline__ v16bf load_a_frag(const __bf16* __restrict__ arow,
                                             int k0, int half) {
  const __bf16* p = arow + k0 + half * 8;
  v8bf lo = *(const v8bf*)(p);
  v8bf hi = *(const v8bf*)(p + 16);
  return cat8(lo, hi);
}

__device__ __forceinline__ v16bf load_b_frag(const __bf16* __restrict__ brow,
                                             int k0, int half) {
  const __bf16* p = brow + k0 + half * 8;
  v8bf lo = *(const v8bf*)(p);
  v8bf hi = *(const v8bf*)(p + 16);
  return cat8(lo, hi);
}

// ---------------------------------------------------------------------------
// GEMM: C[M,128] = act(A[M,128] @ B' + bias), B' stored bf16 [n][k] so that
// B(k,n) = B'[n][k]. One wave computes a 32x16 tile (two 16x16 accumulators
// sharing each B fragment); 8 waves/block span all 128 columns.
// ---------------------------------------------------------------------------
__global__ void gemm_bf16(const __bf16* __restrict__ A,
                          const __bf16* __restrict__ B,
                          const float* __restrict__ bias,
                          __bf16* __restrict__ C,
                          int do_act) {
  const int tileM = blockIdx.x * 32;
  const int wave  = threadIdx.x >> 5;
  const int lane  = threadIdx.x & 31;
  const int half  = lane >> 4;
  const int l15   = lane & 15;
  const int ncol  = wave * 16 + l15;

  const __bf16* arow0 = A + (size_t)(tileM + l15) * D;
  const __bf16* arow1 = arow0 + (size_t)16 * D;
  const __bf16* brow  = B + (size_t)ncol * D;

  v8f c0 = {}, c1 = {};
  #pragma unroll
  for (int kt = 0; kt < 4; ++kt) {
    const int k0 = kt * 32;
    v16bf b  = load_b_frag(brow, k0, half);
    v16bf a0 = load_a_frag(arow0, k0, half);
    v16bf a1 = load_a_frag(arow1, k0, half);
    c0 = __builtin_amdgcn_wmma_f32_16x16x32_bf16(false, a0, false, b, (short)0, c0, false, false);
    c1 = __builtin_amdgcn_wmma_f32_16x16x32_bf16(false, a1, false, b, (short)0, c1, false, false);
  }

  const float bn = bias[ncol];
  #pragma unroll
  for (int r = 0; r < 8; ++r) {
    const int m0 = tileM + r + 8 * half;
    float v0 = c0[r] + bn;
    float v1 = c1[r] + bn;
    if (do_act) { v0 = lrelu(v0); v1 = lrelu(v1); }
    C[(size_t)m0 * D + ncol]        = (__bf16)v0;
    C[(size_t)(m0 + 16) * D + ncol] = (__bf16)v1;
  }
}

// ---------------------------------------------------------------------------
// RGCN combine: out = X@root + mean0@W0 + mean1@W1 + bias.
// All B matrices pre-transposed to [n][k] bf16. 24 WMMA per wave.
// ---------------------------------------------------------------------------
__global__ void gemm_rgcn_bf16(const __bf16* __restrict__ X,
                               const __bf16* __restrict__ M0,
                               const __bf16* __restrict__ M1,
                               const __bf16* __restrict__ RB,
                               const __bf16* __restrict__ W0B,
                               const __bf16* __restrict__ W1B,
                               const float* __restrict__ bias,
                               __bf16* __restrict__ C) {
  const int tileM = blockIdx.x * 32;
  const int wave  = threadIdx.x >> 5;
  const int lane  = threadIdx.x & 31;
  const int half  = lane >> 4;
  const int l15   = lane & 15;
  const int ncol  = wave * 16 + l15;

  const __bf16* As[3] = {X, M0, M1};
  const __bf16* Bs[3] = {RB, W0B, W1B};

  v8f c0 = {}, c1 = {};
  #pragma unroll
  for (int s = 0; s < 3; ++s) {
    const __bf16* arow0 = As[s] + (size_t)(tileM + l15) * D;
    const __bf16* arow1 = arow0 + (size_t)16 * D;
    const __bf16* brow  = Bs[s] + (size_t)ncol * D;
    #pragma unroll
    for (int kt = 0; kt < 4; ++kt) {
      const int k0 = kt * 32;
      v16bf b  = load_b_frag(brow, k0, half);
      v16bf a0 = load_a_frag(arow0, k0, half);
      v16bf a1 = load_a_frag(arow1, k0, half);
      c0 = __builtin_amdgcn_wmma_f32_16x16x32_bf16(false, a0, false, b, (short)0, c0, false, false);
      c1 = __builtin_amdgcn_wmma_f32_16x16x32_bf16(false, a1, false, b, (short)0, c1, false, false);
    }
  }

  const float bn = bias[ncol];
  #pragma unroll
  for (int r = 0; r < 8; ++r) {
    const int m0 = tileM + r + 8 * half;
    C[(size_t)m0 * D + ncol]        = (__bf16)(c0[r] + bn);
    C[(size_t)(m0 + 16) * D + ncol] = (__bf16)(c1[r] + bn);
  }
}

// ---------------------------------------------------------------------------
// Graph scatter: one wave per edge. Feature row moved with packed-bf16
// atomics (global_atomic_pk_add_bf16): 2 dwords (4 bf16) per lane.
// ---------------------------------------------------------------------------
__global__ void zero_u32(uint32_t* __restrict__ p, size_t n) {
  size_t i = (size_t)blockIdx.x * blockDim.x + threadIdx.x;
  if (i < n) p[i] = 0u;
}

__global__ void scatter_edges(const __bf16* __restrict__ X,
                              const int* __restrict__ edge_index,
                              const int* __restrict__ edge_type,
                              __bf16* __restrict__ SUM,
                              float* __restrict__ CNT) {
  const int e = blockIdx.x * 8 + (threadIdx.x >> 5);
  if (e >= N_EDGES) return;
  const int lane = threadIdx.x & 31;
  const int src = edge_index[e];
  const int dst = edge_index[N_EDGES + e];
  const int r   = edge_type[e];
  const uint32_t* __restrict__ xw = (const uint32_t*)(X + (size_t)src * D);
  uint32_t* __restrict__ sw = (uint32_t*)(SUM + ((size_t)r * N_NODES + dst) * D);

  uint32_t d0 = xw[lane];
  uint32_t d1 = xw[lane + 32];
  asm volatile("global_atomic_pk_add_bf16 %0, %1, off"
               :: "v"((uint64_t)(uintptr_t)(sw + lane)), "v"(d0) : "memory");
  asm volatile("global_atomic_pk_add_bf16 %0, %1, off"
               :: "v"((uint64_t)(uintptr_t)(sw + lane + 32)), "v"(d1) : "memory");
  if (lane == 0) atomicAdd(&CNT[(size_t)r * N_NODES + dst], 1.0f);
}

// mean = sum / max(cnt, 1), in place on bf16 rows. One wave per row.
__global__ void divide_rows(__bf16* __restrict__ SUM, const float* __restrict__ CNT) {
  const int row = blockIdx.x * 8 + (threadIdx.x >> 5);
  if (row >= 2 * N_NODES) return;
  const int lane = threadIdx.x & 31;
  const float scale = 1.0f / fmaxf(CNT[row], 1.0f);
  __bf16* __restrict__ s = SUM + (size_t)row * D;
  #pragma unroll
  for (int j = 0; j < 4; ++j) {
    const int i = lane + 32 * j;
    s[i] = (__bf16)((float)s[i] * scale);
  }
}

// ---------------------------------------------------------------------------
// Final head: [200k,128]bf16 x W_o2^T[2,128]f32 + b_o2 -> [200k,2] f32
// ---------------------------------------------------------------------------
__global__ void head2(const __bf16* __restrict__ X,
                      const float* __restrict__ W_o2, const float* __restrict__ b_o2,
                      float* __restrict__ out) {
  const int m = blockIdx.x * blockDim.x + threadIdx.x;
  if (m >= N_NODES) return;
  const __bf16* __restrict__ xr = X + (size_t)m * D;
  float a0 = b_o2[0], a1 = b_o2[1];
  #pragma unroll 4
  for (int k = 0; k < D; ++k) {
    const float v = (float)xr[k];
    a0 += v * W_o2[k];
    a1 += v * W_o2[D + k];
  }
  out[(size_t)m * 2 + 0] = a0;
  out[(size_t)m * 2 + 1] = a1;
}

// ---------------------------------------------------------------------------
extern "C" void kernel_launch(void* const* d_in, const int* in_sizes, int n_in,
                              void* d_out, int out_size, void* d_ws, size_t ws_size,
                              hipStream_t stream) {
  const float* des        = (const float*)d_in[0];
  const float* tweet      = (const float*)d_in[1];
  const float* num_prop   = (const float*)d_in[2];
  const float* cat_prop   = (const float*)d_in[3];
  const int*   edge_index = (const int*)  d_in[4];
  const int*   edge_type  = (const int*)  d_in[5];
  const float* W_des = (const float*)d_in[6];  const float* b_des = (const float*)d_in[7];
  const float* W_num = (const float*)d_in[8];  const float* b_num = (const float*)d_in[9];
  const float* W_cat = (const float*)d_in[10]; const float* b_cat = (const float*)d_in[11];
  const float* W_tw  = (const float*)d_in[12]; const float* b_tw  = (const float*)d_in[13];
  const float* W_in  = (const float*)d_in[14]; const float* b_in  = (const float*)d_in[15];
  const float* rgcn_w    = (const float*)d_in[16];
  const float* rgcn_root = (const float*)d_in[17];
  const float* rgcn_bias = (const float*)d_in[18];
  const float* W_o1 = (const float*)d_in[19];  const float* b_o1 = (const float*)d_in[20];
  const float* W_o2 = (const float*)d_in[21];  const float* b_o2 = (const float*)d_in[22];

  // Workspace layout (bytes):
  //   Xa[200k,128]bf16 | Xb bf16 | SUM[2*200k,128]bf16 | CNT[2*200k]f32 | 5 weight mats bf16
  char* w = (char*)d_ws;
  __bf16* Xa  = (__bf16*)(w);
  __bf16* Xb  = (__bf16*)(w + (size_t)N_NODES * D * 2);
  __bf16* SUM = (__bf16*)(w + (size_t)2 * N_NODES * D * 2);
  float*  CNT = (float*) (w + (size_t)4 * N_NODES * D * 2);
  __bf16* WB  = (__bf16*)(w + (size_t)4 * N_NODES * D * 2 + (size_t)2 * N_NODES * 4);
  __bf16* wb_in = WB;                 // [n][k] == W_in row-major (already [out][in])
  __bf16* wb_o1 = WB + 16384;
  __bf16* rb    = WB + 2 * 16384;     // transpose of rgcn_root
  __bf16* w0b   = WB + 3 * 16384;     // transpose of rgcn_w[0]
  __bf16* w1b   = WB + 4 * 16384;     // transpose of rgcn_w[1]

  // 0) Weight prep (tiny)
  cvt_rowmajor<<<64, 256, 0, stream>>>(W_in, wb_in, 16384);
  cvt_rowmajor<<<64, 256, 0, stream>>>(W_o1, wb_o1, 16384);
  cvt_transpose128<<<64, 256, 0, stream>>>(rgcn_root, rb);
  cvt_transpose128<<<64, 256, 0, stream>>>(rgcn_w, w0b);
  cvt_transpose128<<<64, 256, 0, stream>>>(rgcn_w + (size_t)D * D, w1b);

  // 1) Encoders -> Xa
  encode_users<<<N_USERS, 128, 0, stream>>>(des, num_prop, cat_prop,
                                            W_des, b_des, W_num, b_num, W_cat, b_cat, Xa);
  encode_tweets<<<N_TWEETS, 128, 0, stream>>>(tweet, W_tw, b_tw, Xa + (size_t)N_USERS * D);

  // 2) x = lrelu(x @ W_in^T + b_in) -> Xb
  gemm_bf16<<<N_NODES / 32, 256, 0, stream>>>(Xa, wb_in, b_in, Xb, 1);

  // 3) Two RGCN layers, ping-ponging Xa/Xb
  __bf16* cur = Xb;
  __bf16* nxt = Xa;
  const size_t nz = ((size_t)2 * N_NODES * D * 2 + (size_t)2 * N_NODES * 4) / 4; // SUM+CNT dwords
  for (int layer = 0; layer < 2; ++layer) {
    zero_u32<<<(unsigned)((nz + 255) / 256), 256, 0, stream>>>((uint32_t*)SUM, nz);
    scatter_edges<<<(N_EDGES + 7) / 8, 256, 0, stream>>>(cur, edge_index, edge_type, SUM, CNT);
    divide_rows<<<(2 * N_NODES + 7) / 8, 256, 0, stream>>>(SUM, CNT);
    gemm_rgcn_bf16<<<N_NODES / 32, 256, 0, stream>>>(cur,
                                                     SUM,                        // mean rel 0
                                                     SUM + (size_t)N_NODES * D,  // mean rel 1
                                                     rb, w0b, w1b, rgcn_bias, nxt);
    __bf16* tmp = cur; cur = nxt; nxt = tmp;
  }

  // 4) x = lrelu(x @ W_o1^T + b_o1) -> nxt
  gemm_bf16<<<N_NODES / 32, 256, 0, stream>>>(cur, wb_o1, b_o1, nxt, 1);

  // 5) out = x @ W_o2^T + b_o2 -> d_out [200k, 2]
  head2<<<(N_NODES + 255) / 256, 256, 0, stream>>>(nxt, W_o2, b_o2, (float*)d_out);
}
